// LoopGPT_7069516169352
// MI455X (gfx1250) — compile-verified, matrix operand order checked by reading, not run on hardware
//
#include <hip/hip_runtime.h>
#include <cstdint>
#include <cstddef>

typedef __bf16 bf16_t;
typedef __attribute__((ext_vector_type(16))) __bf16 v16bf;
typedef __attribute__((ext_vector_type(8)))  __bf16 v8bf;
typedef __attribute__((ext_vector_type(8)))  float  v8f;
typedef __attribute__((ext_vector_type(4))) unsigned tdm_g0_t;
typedef __attribute__((ext_vector_type(8))) int      tdm_g1_t;
typedef __attribute__((ext_vector_type(4))) int      tdm_g2_t;
typedef __attribute__((ext_vector_type(8))) int      tdm_g4_t;

#define NB 768
#define TT 512
#define NH 12
#define DHD 64
#define BBATCH 8
#define VV 50257
#define ROWS 4096  // B*T

// ---------- helpers ----------
__device__ __forceinline__ bf16_t f2bf(float f) { return static_cast<bf16_t>(f); }
__device__ __forceinline__ bf16_t bfzero() {
  return __builtin_bit_cast(bf16_t, (unsigned short)0);
}

// Load one 16-bit A/B WMMA fragment from LDS. p points at (row, c0) where
// c0 = kbase + (lane<16 ? 0 : 8). VGPRs 0-3 <- K c0..c0+7, VGPRs 4-7 <- K c0+16..c0+23.
__device__ __forceinline__ v16bf load_frag(const bf16_t* p) {
  v8bf x = *(const v8bf*)p;
  v8bf y = *(const v8bf*)(p + 16);
  v16bf r;
#pragma unroll
  for (int i = 0; i < 8; i++) { r[i] = x[i]; r[i + 8] = y[i]; }
  return r;
}

__device__ __forceinline__ v8f wmma_bf16(v16bf a, v16bf b, v8f c) {
  return __builtin_amdgcn_wmma_f32_16x16x32_bf16(false, a, false, b, (short)0, c,
                                                 false, false);
}

// ---- CDNA5 async global->LDS copy (ASYNCcnt) ----
__device__ __forceinline__ void async_ld_b128(void* lds_dst, const void* gsrc) {
  unsigned lds = (unsigned)(uintptr_t)lds_dst;  // low 32 bits of generic = LDS offset
  asm volatile("global_load_async_to_lds_b128 %0, %1, off"
               :
               : "v"(lds), "v"(gsrc)
               : "memory");
}
__device__ __forceinline__ void wait_async0() {
  asm volatile("s_wait_asynccnt 0x0" ::: "memory");
}

// ---- CDNA5 Tensor Data Mover: 2-D tile (tile_d0 contiguous elems x tile_d1 rows)
// of 2-byte elements, row stride = row_stride_elems, optional LDS row padding.
__device__ __forceinline__ void tdm_load_tile_2d(void* lds_dst, const void* gsrc,
                                                 unsigned tile_d0, unsigned tile_d1,
                                                 unsigned row_stride_elems,
                                                 unsigned pad_interval_code,
                                                 unsigned pad_amount_code,
                                                 unsigned pad_en) {
  unsigned long long ga = (unsigned long long)(uintptr_t)gsrc;
  unsigned lds = (unsigned)(uintptr_t)lds_dst;
  tdm_g0_t g0;
  g0[0] = 1u;                                               // count=1 (valid, user)
  g0[1] = lds;                                              // lds_addr
  g0[2] = (unsigned)ga;                                     // global_addr[31:0]
  g0[3] = (unsigned)((ga >> 32) & 0x1FFFFFFull) | (2u << 30);  // [56:32] | type=2
  tdm_g1_t g1;
  g1[0] = (int)((1u << 16) | (pad_en << 20) |               // data_size=1 (2B)
                (pad_interval_code << 22) | (pad_amount_code << 25));
  g1[1] = (int)((tile_d0 & 0xFFFFu) << 16);                 // tensor_dim0[15:0]
  g1[2] = (int)((tile_d0 >> 16) | ((tile_d1 & 0xFFFFu) << 16));  // dim0 hi | dim1 lo
  g1[3] = (int)((tile_d1 >> 16) | (tile_d0 << 16));         // dim1 hi | tile_dim0
  g1[4] = (int)(tile_d1 & 0xFFFFu);                         // tile_dim1 (tile_dim2=0)
  g1[5] = (int)row_stride_elems;                            // tensor_dim0_stride lo
  g1[6] = 0;
  g1[7] = 0;
  tdm_g2_t gz = (tdm_g2_t)0;
  tdm_g4_t gz8 = (tdm_g4_t)0;
  __builtin_amdgcn_tensor_load_to_lds(g0, g1, gz, gz, gz8, 0);
}

// ---------- weight transpose + fp32->bf16 convert: Wt[n][k] = W[k][n] ----------
__global__ __launch_bounds__(256) void wconv_t(const float* __restrict__ W,
                                               bf16_t* __restrict__ Wt,
                                               int K, int Nc) {
  __shared__ float tile[32][33];
  int n0 = blockIdx.x * 32, k0 = blockIdx.y * 32;
  int tx = threadIdx.x & 31, ty = threadIdx.x >> 5;  // 32 x 8
#pragma unroll
  for (int j = 0; j < 4; j++)
    tile[ty + j * 8][tx] = W[(size_t)(k0 + ty + j * 8) * Nc + n0 + tx];
  __syncthreads();
#pragma unroll
  for (int j = 0; j < 4; j++)
    Wt[(size_t)(n0 + ty + j * 8) * K + k0 + tx] = f2bf(tile[tx][ty + j * 8]);
}

// ---------- embedding: x0 = wte[idx] + wpe ; h = x0 ----------
__global__ __launch_bounds__(256) void embed_k(const int* __restrict__ idx,
                                               const float* __restrict__ wte,
                                               const float* __restrict__ wpe,
                                               float* __restrict__ x0,
                                               float* __restrict__ h) {
  size_t i = (size_t)blockIdx.x * 256 + threadIdx.x;  // ROWS*NB total
  int row = (int)(i / NB), c = (int)(i % NB);
  int t = row & (TT - 1);
  int tok = idx[row];
  float v = wte[(size_t)tok * NB + c] + wpe[(size_t)t * NB + c];
  x0[i] = v;
  h[i] = v;
}

// ---------- layernorm (optionally fused h_in = xin + x0 + se, writes xout) ----------
__global__ __launch_bounds__(256) void ln_k(const float* __restrict__ xin,
                                            const float* __restrict__ x0,
                                            const float* __restrict__ se,
                                            const float* __restrict__ w,
                                            float* __restrict__ xout,
                                            bf16_t* __restrict__ lnout) {
  int row = blockIdx.x, tid = threadIdx.x;
  __shared__ float s1[256], s2[256];
  const size_t base = (size_t)row * NB;
  float v[3];
  float s = 0.f, q = 0.f;
#pragma unroll
  for (int j = 0; j < 3; j++) {
    int c = tid + j * 256;
    float val = xin[base + c];
    if (x0) val += x0[base + c];
    if (se) val += se[c];
    v[j] = val;
    s += val;
    q += val * val;
  }
  s1[tid] = s; s2[tid] = q;
  __syncthreads();
  for (int st = 128; st > 0; st >>= 1) {
    if (tid < st) { s1[tid] += s1[tid + st]; s2[tid] += s2[tid + st]; }
    __syncthreads();
  }
  float mean = s1[0] * (1.0f / NB);
  float var = s2[0] * (1.0f / NB) - mean * mean;
  float rstd = rsqrtf(var + 1e-5f);
#pragma unroll
  for (int j = 0; j < 3; j++) {
    int c = tid + j * 256;
    if (xout) xout[base + c] = v[j];
    lnout[base + c] = f2bf((v[j] - mean) * rstd * w[c]);
  }
}

// ---------- final LN on last position only -> 16-row padded bf16 A matrix ----------
__global__ __launch_bounds__(256) void lnf_k(const float* __restrict__ x,
                                             const float* __restrict__ w,
                                             bf16_t* __restrict__ lmA) {
  int row = blockIdx.x, tid = threadIdx.x;
  if (row >= BBATCH) {
#pragma unroll
    for (int j = 0; j < 3; j++) lmA[(size_t)row * NB + tid + j * 256] = bfzero();
    return;
  }
  __shared__ float s1[256], s2[256];
  const float* xr = x + ((size_t)(row * TT + (TT - 1))) * NB;
  float v[3];
  float s = 0.f, q = 0.f;
#pragma unroll
  for (int j = 0; j < 3; j++) {
    int c = tid + j * 256;
    float val = xr[c];
    v[j] = val; s += val; q += val * val;
  }
  s1[tid] = s; s2[tid] = q;
  __syncthreads();
  for (int st = 128; st > 0; st >>= 1) {
    if (tid < st) { s1[tid] += s1[tid + st]; s2[tid] += s2[tid + st]; }
    __syncthreads();
  }
  float mean = s1[0] * (1.0f / NB);
  float var = s2[0] * (1.0f / NB) - mean * mean;
  float rstd = rsqrtf(var + 1e-5f);
#pragma unroll
  for (int j = 0; j < 3; j++) {
    int c = tid + j * 256;
    lmA[(size_t)row * NB + c] = f2bf((v[j] - mean) * rstd * w[c]);
  }
}

// ---------- GEMM: C[M,Nc] = A[M,K](bf16) x Bt[Nc,K](bf16), TDM double-buffered ----------
// EPI 0: p0[gm*Nc+gn] += val (residual add, fp32)
// EPI 1: o0[gm*Nc+gn] = bf16(gelu(val))
// EPI 2: qkv split -> o0=q(*0.125), o1=k, o2=vT
#define BM 128
#define BN 128
#define BK 32
#define LDT 40  // 64B data + 16B pad per row (TDM: pad_interval=3 -> 64B, pad_amount=3 -> 4 dwords)

template <int EPI>
__device__ __forceinline__ void epi_store(float val, int gm, int gn, int Nc,
                                          float* p0, bf16_t* o0, bf16_t* o1,
                                          bf16_t* o2) {
  if (EPI == 0) {
    size_t o = (size_t)gm * Nc + gn;
    p0[o] += val;
  } else if (EPI == 1) {
    float g = 0.5f * val * (1.0f + erff(val * 0.70710678118f));
    o0[(size_t)gm * Nc + gn] = f2bf(g);
  } else {
    int b = gm >> 9, t = gm & (TT - 1);
    if (gn < NB) {
      int h = gn >> 6, d = gn & 63;
      o0[(((size_t)(b * NH + h) * TT) + t) * DHD + d] = f2bf(val * 0.125f);
    } else if (gn < 2 * NB) {
      int c = gn - NB; int h = c >> 6, d = c & 63;
      o1[(((size_t)(b * NH + h) * TT) + t) * DHD + d] = f2bf(val);
    } else {
      int c = gn - 2 * NB; int h = c >> 6, d = c & 63;
      o2[(((size_t)(b * NH + h) * DHD) + d) * TT + t] = f2bf(val);
    }
  }
}

template <int EPI>
__global__ __launch_bounds__(256) void gemm_bf16(const bf16_t* __restrict__ A,
                                                 const bf16_t* __restrict__ Bt,
                                                 int M, int Nc, int K,
                                                 float* p0, bf16_t* o0,
                                                 bf16_t* o1, bf16_t* o2) {
  __shared__ bf16_t sa[2][BM * LDT];
  __shared__ bf16_t sb[2][BN * LDT];
  int tid = threadIdx.x;
  int lane = tid & 31, wave = tid >> 5;
  int m0 = blockIdx.x * BM, n0 = blockIdx.y * BN;
  int mw = (wave >> 2) * 64;  // 0 or 64
  int nw = (wave & 3) * 32;   // 0..96
  int lr = lane & 15;
  int coff = (lane < 16) ? 0 : 8;

  v8f acc[4][2];
#pragma unroll
  for (int i = 0; i < 4; i++)
#pragma unroll
    for (int j = 0; j < 2; j++) acc[i][j] = (v8f)0.0f;

  const int nk = K / BK;
  auto issue = [&](int it) {
    int k0 = it * BK;
    int buf = it & 1;
    tdm_load_tile_2d(&sa[buf][0], &A[(size_t)m0 * K + k0], BK, BM, K, 3, 3, 1);
    tdm_load_tile_2d(&sb[buf][0], &Bt[(size_t)n0 * K + k0], BK, BN, K, 3, 3, 1);
  };
  if (wave == 0) issue(0);

  for (int it = 0; it < nk; it++) {
    if (wave == 0) {
      if (it + 1 < nk) {
        issue(it + 1);                          // prefetch next tile-pair (DMA)
        __builtin_amdgcn_s_wait_tensorcnt(2);   // current pair landed
      } else {
        __builtin_amdgcn_s_wait_tensorcnt(0);
      }
    }
    __syncthreads();
    int buf = it & 1;
    v16bf af[4], bfr[2];
#pragma unroll
    for (int i = 0; i < 4; i++)
      af[i] = load_frag(&sa[buf][(mw + i * 16 + lr) * LDT + coff]);
#pragma unroll
    for (int j = 0; j < 2; j++)
      bfr[j] = load_frag(&sb[buf][(nw + j * 16 + lr) * LDT + coff]);
#pragma unroll
    for (int i = 0; i < 4; i++)
#pragma unroll
      for (int j = 0; j < 2; j++) acc[i][j] = wmma_bf16(af[i], bfr[j], acc[i][j]);
    __syncthreads();
  }

  int mhalf = (lane < 16) ? 0 : 8;
#pragma unroll
  for (int i = 0; i < 4; i++)
#pragma unroll
    for (int j = 0; j < 2; j++) {
      int gmb = m0 + mw + i * 16 + mhalf;
      int gn = n0 + nw + j * 16 + lr;
#pragma unroll
      for (int v = 0; v < 8; v++)
        epi_store<EPI>(acc[i][j][v], gmb + v, gn, Nc, p0, o0, o1, o2);
    }
}

// ---------- fused causal flash attention (async LDS loads) ----------
// grid: (B*H*4) blocks, 256 threads; block handles 128 q-rows of one head.
#define LDQ 72
#define LDV 136
#define LDP 136

__global__ __launch_bounds__(256) void attn_k(const bf16_t* __restrict__ q,
                                              const bf16_t* __restrict__ k,
                                              const bf16_t* __restrict__ vt,
                                              bf16_t* __restrict__ y) {
  extern __shared__ char dynsm[];
  bf16_t* qs = (bf16_t*)dynsm;            // 128*LDQ
  bf16_t* kss = qs + 128 * LDQ;           // 128*LDQ
  bf16_t* vs = kss + 128 * LDQ;           // 64*LDV
  bf16_t* ps = vs + 64 * LDV;             // 8*16*LDP

  int tid = threadIdx.x, lane = tid & 31, wave = tid >> 5;
  int blk = blockIdx.x;
  int qt = blk & 3, bh = blk >> 2;
  int q0 = qt * 128;
  int b = bh / NH, hh = bh % NH;
  const bf16_t* qg = q + (size_t)bh * TT * DHD;
  const bf16_t* kg = k + (size_t)bh * TT * DHD;
  const bf16_t* vg = vt + (size_t)bh * DHD * TT;

  {  // Q tile [128][64] via async global->LDS
    int row = tid >> 1, half = tid & 1;
#pragma unroll
    for (int j = 0; j < 4; j++)
      async_ld_b128(&qs[row * LDQ + half * 32 + j * 8],
                    &qg[(size_t)(q0 + row) * DHD + half * 32 + j * 8]);
  }
  wait_async0();
  __syncthreads();

  int lr = lane & 15;
  int coff = (lane < 16) ? 0 : 8;
  int mhalf = (lane < 16) ? 0 : 8;
  int qrow0 = wave * 16;

  v16bf qa[2];
  qa[0] = load_frag(&qs[(qrow0 + lr) * LDQ + 0 + coff]);
  qa[1] = load_frag(&qs[(qrow0 + lr) * LDQ + 32 + coff]);

  float m_run[8], l_run[8];
  v8f oacc[4];
#pragma unroll
  for (int v = 0; v < 8; v++) { m_run[v] = -1e30f; l_run[v] = 0.0f; }
#pragma unroll
  for (int j = 0; j < 4; j++) oacc[j] = (v8f)0.0f;

  for (int kt = 0; kt <= qt; kt++) {
    int k0 = kt * 128;
    __syncthreads();  // previous-iter LDS consumers done
    {  // K tile [128][64] async
      int row = tid >> 1, half = tid & 1;
#pragma unroll
      for (int j = 0; j < 4; j++)
        async_ld_b128(&kss[row * LDQ + half * 32 + j * 8],
                      &kg[(size_t)(k0 + row) * DHD + half * 32 + j * 8]);
    }
    {  // Vt tile [64][128] async
      int row = tid >> 2, seg = (tid & 3) * 32;
#pragma unroll
      for (int j = 0; j < 4; j++)
        async_ld_b128(&vs[row * LDV + seg + j * 8],
                      &vg[(size_t)row * TT + k0 + seg + j * 8]);
    }
    wait_async0();
    __syncthreads();

    v8f sacc[8];
#pragma unroll
    for (int nt = 0; nt < 8; nt++) sacc[nt] = (v8f)0.0f;
#pragma unroll
    for (int ks = 0; ks < 2; ks++) {
#pragma unroll
      for (int nt = 0; nt < 8; nt++) {
        v16bf kb = load_frag(&kss[(nt * 16 + lr) * LDQ + ks * 32 + coff]);
        sacc[nt] = wmma_bf16(qa[ks], kb, sacc[nt]);
      }
    }

    if (kt == qt) {  // diagonal tile: causal mask (q0 == k0)
#pragma unroll
      for (int nt = 0; nt < 8; nt++)
#pragma unroll
        for (int v = 0; v < 8; v++) {
          int qi = qrow0 + v + mhalf;
          int ki = nt * 16 + lr;
          if (ki > qi) sacc[nt][v] = -1e30f;
        }
    }

    float scale_old[8];
#pragma unroll
    for (int v = 0; v < 8; v++) {
      float mx = sacc[0][v];
#pragma unroll
      for (int nt = 1; nt < 8; nt++) mx = fmaxf(mx, sacc[nt][v]);
      mx = fmaxf(mx, __shfl_xor(mx, 1, 32));
      mx = fmaxf(mx, __shfl_xor(mx, 2, 32));
      mx = fmaxf(mx, __shfl_xor(mx, 4, 32));
      mx = fmaxf(mx, __shfl_xor(mx, 8, 32));
      float mn = fmaxf(m_run[v], mx);
      float so = __expf(m_run[v] - mn);
      float ls = 0.0f;
#pragma unroll
      for (int nt = 0; nt < 8; nt++) {
        float p = __expf(sacc[nt][v] - mn);
        sacc[nt][v] = p;
        ls += p;
      }
      ls += __shfl_xor(ls, 1, 32);
      ls += __shfl_xor(ls, 2, 32);
      ls += __shfl_xor(ls, 4, 32);
      ls += __shfl_xor(ls, 8, 32);
      l_run[v] = l_run[v] * so + ls;
      m_run[v] = mn;
      scale_old[v] = so;
    }
#pragma unroll
    for (int j = 0; j < 4; j++)
#pragma unroll
      for (int v = 0; v < 8; v++) oacc[j][v] *= scale_old[v];

    // C-layout P -> per-wave LDS (bf16), then reload as A fragments
    bf16_t* pw = ps + wave * 16 * LDP;
#pragma unroll
    for (int nt = 0; nt < 8; nt++)
#pragma unroll
      for (int v = 0; v < 8; v++)
        pw[(v + mhalf) * LDP + nt * 16 + lr] = f2bf(sacc[nt][v]);
    __syncthreads();

#pragma unroll
    for (int ks = 0; ks < 4; ks++) {
      v16bf pa = load_frag(&pw[lr * LDP + ks * 32 + coff]);
#pragma unroll
      for (int j = 0; j < 4; j++) {
        v16bf vb = load_frag(&vs[(j * 16 + lr) * LDV + ks * 32 + coff]);
        oacc[j] = wmma_bf16(pa, vb, oacc[j]);
      }
    }
  }

#pragma unroll
  for (int j = 0; j < 4; j++)
#pragma unroll
    for (int v = 0; v < 8; v++) {
      int t = q0 + qrow0 + v + mhalf;
      float val = oacc[j][v] / l_run[v];
      y[((size_t)(b * TT + t)) * NB + hh * DHD + j * 16 + lr] = f2bf(val);
    }
}

// ---------- lm head: out[8,V] = lmA[16,768](bf16) x wte^T, wte cvt on the fly ----------
#define LDA_LM 776
__global__ __launch_bounds__(256) void lmhead_k(const bf16_t* __restrict__ lmA,
                                                const float* __restrict__ wte,
                                                float* __restrict__ out) {
  __shared__ bf16_t sA[16 * LDA_LM];
  __shared__ bf16_t sB[128 * LDT];
  int tid = threadIdx.x, lane = tid & 31, wave = tid >> 5;
  int n0 = blockIdx.x * 128;
  int lr = lane & 15, coff = (lane < 16) ? 0 : 8;

  {  // stage whole A [16][768]
    int row = tid >> 4, seg = (tid & 15) * 48;
#pragma unroll
    for (int j = 0; j < 6; j++)
      *(uint4*)&sA[row * LDA_LM + seg + j * 8] =
          *(const uint4*)&lmA[(size_t)row * NB + seg + j * 8];
  }

  v8f acc = (v8f)0.0f;
  for (int k0 = 0; k0 < NB; k0 += 32) {
    __syncthreads();
    {  // Bt tile: wte rows [n0..n0+127][k0..k0+31], fp32 -> bf16
      int n = tid >> 1, half = (tid & 1) * 16;
      int gn = n0 + n;
      __align__(16) bf16_t tmp[16];
      if (gn < VV) {
#pragma unroll
        for (int i = 0; i < 16; i++)
          tmp[i] = f2bf(wte[(size_t)gn * NB + k0 + half + i]);
      } else {
#pragma unroll
        for (int i = 0; i < 16; i++) tmp[i] = bfzero();
      }
      *(uint4*)&sB[n * LDT + half] = *(uint4*)&tmp[0];
      *(uint4*)&sB[n * LDT + half + 8] = *(uint4*)&tmp[8];
    }
    __syncthreads();
    v16bf af = load_frag(&sA[lr * LDA_LM + k0 + coff]);
    v16bf bfr = load_frag(&sB[(wave * 16 + lr) * LDT + coff]);
    acc = wmma_bf16(af, bfr, acc);
  }

  if (lane < 16) {  // rows 0..7 (valid batches) live in lanes 0..15
    int gn = n0 + wave * 16 + lr;
    if (gn < VV) {
#pragma unroll
      for (int v = 0; v < 8; v++) out[(size_t)v * VV + gn] = acc[v];
    }
  }
}

// ---------- host orchestration ----------
extern "C" void kernel_launch(void* const* d_in, const int* in_sizes, int n_in,
                              void* d_out, int out_size, void* d_ws,
                              size_t ws_size, hipStream_t stream) {
  const int* idx = (const int*)d_in[0];
  const float* wte = (const float*)d_in[1];
  const float* wpe = (const float*)d_in[2];
  const float* step_emb = (const float*)d_in[3];
  const float* ln1_w = (const float*)d_in[4];
  const float* ln2_w = (const float*)d_in[5];
  const float* lnf_w = (const float*)d_in[6];
  const float* c_attn = (const float*)d_in[7];
  const float* attn_proj = (const float*)d_in[8];
  const float* c_fc = (const float*)d_in[9];
  const float* mlp_proj = (const float*)d_in[10];
  float* out = (float*)d_out;
  char* ws = (char*)d_ws;

  size_t off = 0;
  auto alloc = [&](size_t bytes) {
    size_t o = off;
    off += (bytes + 255) & ~(size_t)255;
    return o;
  };
  float* x0 = (float*)(ws + alloc((size_t)ROWS * NB * 4));
  float* xbuf = (float*)(ws + alloc((size_t)ROWS * NB * 4));
  bf16_t* lnb = (bf16_t*)(ws + alloc((size_t)ROWS * NB * 2));
  bf16_t* qb = (bf16_t*)(ws + alloc((size_t)ROWS * NB * 2));
  bf16_t* kb = (bf16_t*)(ws + alloc((size_t)ROWS * NB * 2));
  bf16_t* vtb = (bf16_t*)(ws + alloc((size_t)ROWS * NB * 2));
  bf16_t* yb = (bf16_t*)(ws + alloc((size_t)ROWS * NB * 2));
  bf16_t* actb = (bf16_t*)(ws + alloc((size_t)ROWS * 4 * NB * 2));
  bf16_t* lmA = (bf16_t*)(ws + alloc((size_t)16 * NB * 2));
  bf16_t* wt_attn = (bf16_t*)(ws + alloc((size_t)3 * NB * NB * 2));
  bf16_t* wt_aproj = (bf16_t*)(ws + alloc((size_t)NB * NB * 2));
  bf16_t* wt_fc = (bf16_t*)(ws + alloc((size_t)4 * NB * NB * 2));
  bf16_t* wt_mproj = (bf16_t*)(ws + alloc((size_t)4 * NB * NB * 2));

  // one-time weight transpose+convert (Wt[n][k] = bf16(W[k][n]))
  wconv_t<<<dim3((3 * NB) / 32, NB / 32), 256, 0, stream>>>(c_attn, wt_attn, NB, 3 * NB);
  wconv_t<<<dim3(NB / 32, NB / 32), 256, 0, stream>>>(attn_proj, wt_aproj, NB, NB);
  wconv_t<<<dim3((4 * NB) / 32, NB / 32), 256, 0, stream>>>(c_fc, wt_fc, NB, 4 * NB);
  wconv_t<<<dim3(NB / 32, (4 * NB) / 32), 256, 0, stream>>>(mlp_proj, wt_mproj, 4 * NB, NB);

  embed_k<<<(ROWS * NB) / 256, 256, 0, stream>>>(idx, wte, wpe, x0, xbuf);

  const size_t attn_smem = (size_t)(128 * LDQ + 128 * LDQ + 64 * LDV + 8 * 16 * LDP) * 2;

  for (int l = 0; l < 12; l++) {
    // h_in = h + x0 + se[l]; xbuf <- h_in; lnb <- bf16(LN(h_in)*w1)
    ln_k<<<ROWS, 256, 0, stream>>>(xbuf, x0, step_emb + (size_t)l * NB, ln1_w, xbuf, lnb);
    // qkv = ln @ c_attn, split/scale/transpose-V
    gemm_bf16<2><<<dim3(ROWS / BM, (3 * NB) / BN), 256, 0, stream>>>(
        lnb, wt_attn, ROWS, 3 * NB, NB, nullptr, qb, kb, vtb);
    // causal flash attention
    attn_k<<<BBATCH * NH * (TT / 128), 256, attn_smem, stream>>>(qb, kb, vtb, yb);
    // x += y @ attn_proj
    gemm_bf16<0><<<dim3(ROWS / BM, NB / BN), 256, 0, stream>>>(
        yb, wt_aproj, ROWS, NB, NB, xbuf, nullptr, nullptr, nullptr);
    // lnb <- bf16(LN(x)*w2)
    ln_k<<<ROWS, 256, 0, stream>>>(xbuf, nullptr, nullptr, ln2_w, nullptr, lnb);
    // act = gelu(ln @ c_fc)
    gemm_bf16<1><<<dim3(ROWS / BM, (4 * NB) / BN), 256, 0, stream>>>(
        lnb, wt_fc, ROWS, 4 * NB, NB, nullptr, actb, nullptr, nullptr);
    // x += act @ mlp_proj
    gemm_bf16<0><<<dim3(ROWS / BM, NB / BN), 256, 0, stream>>>(
        actb, wt_mproj, ROWS, NB, 4 * NB, xbuf, nullptr, nullptr, nullptr);
  }

  // final LN on last position -> padded [16,768] bf16, then logits
  lnf_k<<<16, 256, 0, stream>>>(xbuf, lnf_w, lmA);
  lmhead_k<<<(VV + 127) / 128, 256, 0, stream>>>(lmA, wte, out);
}
